// ExpertsChooseBlock_56487409877316
// MI455X (gfx1250) — compile-verified, matrix operand order checked by reading, not run on hardware
//
#include <hip/hip_runtime.h>

// ---------------- types & helpers ----------------
typedef __attribute__((ext_vector_type(16))) __bf16 v16bf;
typedef __attribute__((ext_vector_type(8)))  __bf16 bf16x8;
typedef __attribute__((ext_vector_type(8)))  float  v8f;
typedef int v4i __attribute__((vector_size(16)));

#define DEV __device__ __forceinline__

// gfx1250 async global->LDS copy path (ASYNCcnt), guarded so compile never breaks.
#if defined(__has_builtin)
#if __has_builtin(__builtin_amdgcn_global_load_async_to_lds_b128) && \
    __has_builtin(__builtin_amdgcn_s_wait_asynccnt)
#define USE_ASYNC_LDS 1
#endif
#endif

DEV void copy16_g2s(__bf16* dst_lds, const __bf16* src_glob) {
#ifdef USE_ASYNC_LDS
    __builtin_amdgcn_global_load_async_to_lds_b128(
        (__attribute__((address_space(1))) v4i*)(void*)const_cast<__bf16*>(src_glob),
        (__attribute__((address_space(3))) v4i*)(void*)dst_lds, 0, 0);
#else
    *(bf16x8*)dst_lds = *(const bf16x8*)src_glob;
#endif
}
DEV void g2s_commit() {
#ifdef USE_ASYNC_LDS
    __builtin_amdgcn_s_wait_asynccnt(0);
#endif
}

DEV __bf16 f2bf(float f) {
    unsigned u = __float_as_uint(f);
    unsigned r = (u + 0x7FFFu + ((u >> 16) & 1u)) >> 16;
    unsigned short s = (unsigned short)r;
    return __builtin_bit_cast(__bf16, s);
}
DEV float bf2f(__bf16 b) {
    unsigned short s = __builtin_bit_cast(unsigned short, b);
    return __uint_as_float(((unsigned)s) << 16);
}
DEV float gelu_tanh(float x) {
    const float c0 = 0.7978845608028654f, c1 = 0.044715f;
    float t = tanhf(c0 * (x + c1 * x * x * x));
    return 0.5f * x * (1.f + t);
}
// Load one 16x16x32 bf16 WMMA fragment row-slice: lane owns row `row`,
// elements 0..7 <- K[kb+8g .. +7], 8..15 <- K[kb+16+8g .. +7] (ISA 7.12.2 layout).
DEV v16bf load_frag(const __bf16* row, int kb, int grp) {
    bf16x8 lo = *(const bf16x8*)(row + kb + 8 * grp);
    bf16x8 hi = *(const bf16x8*)(row + kb + 16 + 8 * grp);
    v16bf f;
#pragma unroll
    for (int j = 0; j < 8; ++j) { f[j] = lo[j]; f[8 + j] = hi[j]; }
    return f;
}

// ---------------- problem constants ----------------
#define BB   8
#define NN   2048
#define DD   768
#define EE   4
#define HH   12
#define HID  3072
#define CAPC 512
#define DHH  64
#define ZZ   (BB * EE)     // 32
#define Z2   (ZZ * HH)     // 384

// ---------------- small utility kernels ----------------
__global__ void convert_bf16_kernel(const float* __restrict__ src, __bf16* __restrict__ dst, long n) {
    for (long i = (long)blockIdx.x * blockDim.x + threadIdx.x; i < n; i += (long)gridDim.x * blockDim.x)
        dst[i] = f2bf(src[i]);
}
__global__ void copy_f32_kernel(const float* __restrict__ src, float* __restrict__ dst, long n) {
    for (long i = (long)blockIdx.x * blockDim.x + threadIdx.x; i < n; i += (long)gridDim.x * blockDim.x)
        dst[i] = src[i];
}

// ---------------- router: probs[b][e][n] ----------------
__global__ __launch_bounds__(256) void router_kernel(const float* __restrict__ x,
                                                     const float* __restrict__ Wr,
                                                     float* __restrict__ probs) {
    int tok = blockIdx.x * 8 + (threadIdx.x >> 5);
    int lane = threadIdx.x & 31;
    if (tok >= BB * NN) return;
    const float* xr = x + (long)tok * DD;
    float a0 = 0.f, a1 = 0.f, a2 = 0.f, a3 = 0.f;
    for (int i = lane; i < DD; i += 32) {
        float xv = xr[i];
        a0 += xv * Wr[i * EE + 0];
        a1 += xv * Wr[i * EE + 1];
        a2 += xv * Wr[i * EE + 2];
        a3 += xv * Wr[i * EE + 3];
    }
#pragma unroll
    for (int off = 16; off; off >>= 1) {
        a0 += __shfl_xor(a0, off, 32);
        a1 += __shfl_xor(a1, off, 32);
        a2 += __shfl_xor(a2, off, 32);
        a3 += __shfl_xor(a3, off, 32);
    }
    if (lane == 0) {
        float mx = fmaxf(fmaxf(a0, a1), fmaxf(a2, a3));
        float e0 = __expf(a0 - mx), e1 = __expf(a1 - mx), e2 = __expf(a2 - mx), e3 = __expf(a3 - mx);
        float inv = 1.f / (e0 + e1 + e2 + e3);
        int b = tok / NN, n = tok % NN;
        probs[((long)b * EE + 0) * NN + n] = e0 * inv;
        probs[((long)b * EE + 1) * NN + n] = e1 * inv;
        probs[((long)b * EE + 2) * NN + n] = e2 * inv;
        probs[((long)b * EE + 3) * NN + n] = e3 * inv;
    }
}

// ---------------- exact top-CAP per (b,e): binary search on float bits ----------------
__global__ __launch_bounds__(256) void topk_kernel(const float* __restrict__ probs,
                                                   int* __restrict__ idxb,
                                                   float* __restrict__ gateb) {
    int z = blockIdx.x;                    // 0..31
    const float* P = probs + (long)z * NN;
    __shared__ int s_cnt;
    __shared__ unsigned s_lo, s_hi;
    __shared__ int s_nsel;
    int tid = threadIdx.x;
    if (tid == 0) { s_lo = 0u; s_hi = 0x7F800000u; }
    __syncthreads();
    for (int it = 0; it < 31; ++it) {
        unsigned lo = s_lo, hi = s_hi;
        if (hi - lo <= 1u) break;          // uniform (shared state)
        unsigned mid = lo + ((hi - lo) >> 1);
        if (tid == 0) s_cnt = 0;
        __syncthreads();
        int c = 0;
        for (int i = tid; i < NN; i += 256)
            if (__float_as_uint(P[i]) >= mid) ++c;
        atomicAdd(&s_cnt, c);
        __syncthreads();
        if (tid == 0) { if (s_cnt >= CAPC) s_lo = mid; else s_hi = mid; }
        __syncthreads();
    }
    unsigned thr = s_lo;                   // cnt(>=thr) >= CAP, cnt(>thr) < CAP
    if (tid == 0) s_nsel = 0;
    __syncthreads();
    for (int i = tid; i < NN; i += 256) {
        if (__float_as_uint(P[i]) > thr) {
            int p = atomicAdd(&s_nsel, 1);
            if (p < CAPC) { idxb[(long)z * CAPC + p] = i; gateb[(long)z * CAPC + p] = P[i]; }
        }
    }
    __syncthreads();
    for (int i = tid; i < NN; i += 256) {
        if (__float_as_uint(P[i]) == thr) {
            int p = atomicAdd(&s_nsel, 1);
            if (p < CAPC) { idxb[(long)z * CAPC + p] = i; gateb[(long)z * CAPC + p] = P[i]; }
        }
    }
}

// ---------------- gather + LayerNorm -> bf16 expert tiles ----------------
__global__ __launch_bounds__(256) void gather_ln_kernel(const float* __restrict__ src,
                                                        const int* __restrict__ idxb,
                                                        const float* __restrict__ g,
                                                        const float* __restrict__ bta,
                                                        __bf16* __restrict__ out) {
    int c = blockIdx.x, z = blockIdx.y;
    int t = idxb[(long)z * CAPC + c];
    int b = z >> 2;
    const float* xr = src + ((long)b * NN + t) * DD;
    __shared__ float r1[256], r2[256];
    int tid = threadIdx.x;
    float s1 = 0.f, s2 = 0.f;
    for (int i = tid; i < DD; i += 256) { float v = xr[i]; s1 += v; s2 += v * v; }
    r1[tid] = s1; r2[tid] = s2;
    __syncthreads();
    for (int off = 128; off; off >>= 1) {
        if (tid < off) { r1[tid] += r1[tid + off]; r2[tid] += r2[tid + off]; }
        __syncthreads();
    }
    float mu = r1[0] * (1.f / DD);
    float var = r2[0] * (1.f / DD) - mu * mu;
    float rs = rsqrtf(var + 1e-5f);
    __bf16* o = out + ((long)z * CAPC + c) * DD;
    for (int i = tid; i < DD; i += 256) o[i] = f2bf((xr[i] - mu) * rs * g[i] + bta[i]);
}

// ---------------- generic WMMA bf16 GEMM, 128x128x32 tiles, double-buffered ----------------
// C[z] (MxN, bf16) = A[z] (MxKbase, use K<Klim) * B (KbasexN shared), per-expert
// Klim = Kbase>>e, column mask (n % colmod) < (Nbase>>e), e = z % 4.
// EPI: 0 = mask only, 1 = bias+mask, 2 = bias+gelu+mask.
template <int EPI>
__global__ __launch_bounds__(256) void gemm_bf16_kernel(const __bf16* __restrict__ A,
                                                        const __bf16* __restrict__ Bm,
                                                        const float* __restrict__ bias,
                                                        __bf16* __restrict__ Cout,
                                                        int M, int N, int Kbase, int Nbase, int colmod) {
    __shared__ __attribute__((aligned(16))) __bf16 sA[2][128][40];  // pad -> conflict-free, 16B rows
    __shared__ __attribute__((aligned(16))) __bf16 sB[2][128][40];  // B transposed: [n][k]
    int z = blockIdx.z;
    int e = z & 3;
    int Klim = Kbase >> e;
    int Nlim = Nbase >> e;
    const __bf16* Az = A + (long)z * M * Kbase;
    __bf16* Cz = Cout + (long)z * M * N;
    int n0 = blockIdx.x * 128, m0 = blockIdx.y * 128;
    int tid = threadIdx.x, lane = tid & 31, wave = tid >> 5;
    int wm = wave >> 2, wn = wave & 3;     // 2x4 wave grid: 64x32 per wave
    int grp = lane >> 4, lr = lane & 15;
    v8f vzero = {0.f, 0.f, 0.f, 0.f, 0.f, 0.f, 0.f, 0.f};
    v8f acc[4][2];
#pragma unroll
    for (int mi = 0; mi < 4; ++mi)
#pragma unroll
        for (int ni = 0; ni < 2; ++ni) acc[mi][ni] = vzero;

    auto stage = [&](int buf, int k0) {
        // A tile 128x32 (row-major) via async global->LDS
        for (int c = tid; c < 512; c += 256) {
            int r = c >> 2, kk = (c & 3) << 3;
            copy16_g2s(&sA[buf][r][kk], Az + (long)(m0 + r) * Kbase + k0 + kk);
        }
        // B tile 32x128, transposed into [n][k] (register transpose)
        for (int c = tid; c < 512; c += 256) {
            int kr = c >> 4, nc = (c & 15) << 3;
            bf16x8 vv = *(const bf16x8*)(Bm + (long)(k0 + kr) * N + n0 + nc);
#pragma unroll
            for (int j = 0; j < 8; ++j) sB[buf][nc + j][kr] = vv[j];
        }
    };

    stage(0, 0);
    g2s_commit();
    __syncthreads();
    int buf = 0;
    for (int k0 = 0; k0 < Klim; k0 += 32) {
        int nxt = buf ^ 1;
        if (k0 + 32 < Klim) stage(nxt, k0 + 32);   // prefetch next K-tile, overlaps WMMA below
        v16bf afrag[4], bfrag[2];
#pragma unroll
        for (int mi = 0; mi < 4; ++mi)
            afrag[mi] = load_frag(&sA[buf][wm * 64 + mi * 16 + lr][0], 0, grp);
#pragma unroll
        for (int ni = 0; ni < 2; ++ni)
            bfrag[ni] = load_frag(&sB[buf][wn * 32 + ni * 16 + lr][0], 0, grp);
#pragma unroll
        for (int mi = 0; mi < 4; ++mi)
#pragma unroll
            for (int ni = 0; ni < 2; ++ni)
                acc[mi][ni] = __builtin_amdgcn_wmma_f32_16x16x32_bf16(
                    false, afrag[mi], false, bfrag[ni], (short)0, acc[mi][ni], false, false);
        g2s_commit();
        __syncthreads();
        buf = nxt;
    }
    // epilogue
#pragma unroll
    for (int mi = 0; mi < 4; ++mi) {
#pragma unroll
        for (int ni = 0; ni < 2; ++ni) {
            int gn = n0 + wn * 32 + ni * 16 + lr;
            bool valid = (gn % colmod) < Nlim;
            float bv = (EPI >= 1) ? bias[gn] : 0.f;
#pragma unroll
            for (int v = 0; v < 8; ++v) {
                int gm = m0 + wm * 64 + mi * 16 + v + 8 * grp;
                float val = acc[mi][ni][v] + bv;
                if (EPI == 2) val = gelu_tanh(val);
                if (!valid) val = 0.f;
                Cz[(long)gm * N + gn] = f2bf(val);
            }
        }
    }
}

// ---------------- reorders ----------------
__global__ void reorder_qkv_kernel(const __bf16* __restrict__ qkv, __bf16* __restrict__ qh,
                                   __bf16* __restrict__ kh, __bf16* __restrict__ vh) {
    long total = (long)ZZ * CAPC * (3 * DD);
    for (long i = (long)blockIdx.x * blockDim.x + threadIdx.x; i < total; i += (long)gridDim.x * blockDim.x) {
        long zc = i / (3 * DD);
        int col = (int)(i % (3 * DD));
        int z = (int)(zc / CAPC), c = (int)(zc % CAPC);
        int seg = col / DD, h = (col % DD) / DHH, d = col % DHH;
        long dst = (((long)z * HH + h) * CAPC + c) * DHH + d;
        __bf16* dp = (seg == 0) ? qh : (seg == 1) ? kh : vh;
        dp[dst] = qkv[i];
    }
}
__global__ void reorder_o_kernel(const __bf16* __restrict__ oh, __bf16* __restrict__ of) {
    long total = (long)ZZ * CAPC * DD;
    for (long i = (long)blockIdx.x * blockDim.x + threadIdx.x; i < total; i += (long)gridDim.x * blockDim.x) {
        long zc = i / DD;
        int col = (int)(i % DD);
        int z = (int)(zc / CAPC), c = (int)(zc % CAPC);
        int h = col / DHH, d = col % DHH;
        of[i] = oh[(((long)z * HH + h) * CAPC + c) * DHH + d];
    }
}

// ---------------- fused attention: per (z2, 64-row q tile) ----------------
// LDS: sS 64x520 f32 (133120 B) | sQ 64x72 bf16 (9216 B) | sKV 64x72 bf16 (9216 B)
#define SROW 520
#define QROW 72
#define SMEM_ATTN (64 * SROW * 4 + 2 * 64 * QROW * 2)

__global__ __launch_bounds__(256) void attn_fused_kernel(const __bf16* __restrict__ qh,
                                                         const __bf16* __restrict__ kh,
                                                         const __bf16* __restrict__ vh,
                                                         __bf16* __restrict__ oh) {
    extern __shared__ char smem[];
    float* sS = (float*)smem;
    __bf16* sQ = (__bf16*)(smem + 64 * SROW * 4);
    __bf16* sKV = sQ + 64 * QROW;
    int z2 = blockIdx.y, m0 = blockIdx.x * 64;
    const __bf16* Q = qh + (long)z2 * CAPC * DHH;
    const __bf16* K = kh + (long)z2 * CAPC * DHH;
    const __bf16* V = vh + (long)z2 * CAPC * DHH;
    __bf16* O = oh + (long)z2 * CAPC * DHH;
    int tid = threadIdx.x, lane = tid & 31, wave = tid >> 5;
    int grp = lane >> 4, lr = lane & 15;
    int strip = (wave & 3) * 16;       // M sub-strip
    int half = (wave >> 2) * 32;       // N half
    v8f vzero = {0.f, 0.f, 0.f, 0.f, 0.f, 0.f, 0.f, 0.f};

    // load Q tile (64 x 64), async
    for (int c = tid; c < 512; c += 256) {
        int r = c >> 3, cc = (c & 7) << 3;
        copy16_g2s(&sQ[r * QROW + cc], Q + (long)(m0 + r) * DHH + cc);
    }
    // S = Q * K^T  (64 x 512)
    for (int kt = 0; kt < 8; ++kt) {
        __syncthreads();
        for (int c = tid; c < 512; c += 256) {
            int r = c >> 3, cc = (c & 7) << 3;
            copy16_g2s(&sKV[r * QROW + cc], K + (long)(kt * 64 + r) * DHH + cc);
        }
        g2s_commit();   // also covers the Q-tile async loads on first iteration
        __syncthreads();
#pragma unroll
        for (int ni = 0; ni < 2; ++ni) {
            v8f accS = vzero;
#pragma unroll
            for (int ks = 0; ks < 2; ++ks) {
                v16bf af = load_frag(&sQ[(strip + lr) * QROW], ks * 32, grp);
                v16bf bfr = load_frag(&sKV[(half + ni * 16 + lr) * QROW], ks * 32, grp);
                accS = __builtin_amdgcn_wmma_f32_16x16x32_bf16(false, af, false, bfr, (short)0, accS, false, false);
            }
#pragma unroll
            for (int v = 0; v < 8; ++v) {
                int m = strip + v + 8 * grp;
                int n = kt * 64 + half + ni * 16 + lr;
                sS[m * SROW + n] = accS[v];
            }
        }
    }
    __syncthreads();
    // softmax over rows (scale 1/sqrt(64)), P written in-place as bf16
    if (tid < 64) {
        float* row = sS + tid * SROW;
        float mx = -3.4e38f;
        for (int c = 0; c < 512; ++c) mx = fmaxf(mx, row[c]);
        float sum = 0.f;
        const float sc = 0.125f;
        for (int c = 0; c < 512; ++c) { float ev = __expf((row[c] - mx) * sc); row[c] = ev; sum += ev; }
        float inv = 1.f / sum;
        __bf16* prow = (__bf16*)row;
        for (int c = 0; c < 512; ++c) prow[c] = f2bf(row[c] * inv);  // write@2c <= read@4c: safe ascending
    }
    __syncthreads();
    // O = P * V  (64 x 64), V staged transposed: sKV[d][t]
    v8f accO[2];
    accO[0] = vzero; accO[1] = vzero;
    for (int vt = 0; vt < 8; ++vt) {
        __syncthreads();
        for (int c = tid; c < 512; c += 256) {
            int r = c >> 3, cc = (c & 7) << 3;
            bf16x8 vv = *(const bf16x8*)(V + (long)(vt * 64 + r) * DHH + cc);
#pragma unroll
            for (int j = 0; j < 8; ++j) sKV[(cc + j) * QROW + r] = vv[j];
        }
        __syncthreads();
#pragma unroll
        for (int ks = 0; ks < 2; ++ks) {
            const __bf16* pr = (const __bf16*)(sS + (strip + lr) * SROW);
            v16bf af = load_frag(pr, vt * 64 + ks * 32, grp);
#pragma unroll
            for (int ni = 0; ni < 2; ++ni) {
                v16bf bfr = load_frag(&sKV[(half + ni * 16 + lr) * QROW], ks * 32, grp);
                accO[ni] = __builtin_amdgcn_wmma_f32_16x16x32_bf16(false, af, false, bfr, (short)0, accO[ni], false, false);
            }
        }
    }
#pragma unroll
    for (int ni = 0; ni < 2; ++ni)
#pragma unroll
        for (int v = 0; v < 8; ++v) {
            int m = m0 + strip + v + 8 * grp;
            int n = half + ni * 16 + lr;
            O[(long)m * DHH + n] = f2bf(accO[ni][v]);
        }
}

// ---------------- gated scatter-add back into residual ----------------
__global__ __launch_bounds__(256) void scatter_kernel(const __bf16* __restrict__ ye,
                                                      const int* __restrict__ idxb,
                                                      const float* __restrict__ gateb,
                                                      float* __restrict__ out) {
    int c = blockIdx.x, z = blockIdx.y;
    int t = idxb[(long)z * CAPC + c];
    float gv = gateb[(long)z * CAPC + c];
    int b = z >> 2;
    const __bf16* yr = ye + ((long)z * CAPC + c) * DD;
    float* orow = out + ((long)b * NN + t) * DD;
    for (int i = threadIdx.x; i < DD; i += 256)
        atomicAdd(&orow[i], gv * bf2f(yr[i]));
}

// ---------------- host orchestration ----------------
extern "C" void kernel_launch(void* const* d_in, const int* in_sizes, int n_in,
                              void* d_out, int out_size, void* d_ws, size_t ws_size,
                              hipStream_t stream) {
    const float* x     = (const float*)d_in[0];
    const float* Wr    = (const float*)d_in[1];
    const float* ln1g  = (const float*)d_in[2];
    const float* ln1b  = (const float*)d_in[3];
    const float* ln2g  = (const float*)d_in[4];
    const float* ln2b  = (const float*)d_in[5];
    const float* Wqkv  = (const float*)d_in[6];
    const float* Wproj = (const float*)d_in[7];
    const float* bproj = (const float*)d_in[8];
    const float* W1    = (const float*)d_in[9];
    const float* b1    = (const float*)d_in[10];
    const float* W2    = (const float*)d_in[11];
    const float* b2    = (const float*)d_in[12];
    float* out = (float*)d_out;

    char* w = (char*)d_ws;
    auto take = [&](size_t bytes) -> char* {
        char* p = w;
        w += (bytes + 255) & ~(size_t)255;
        return p;
    };
    float*  probs  = (float*)take((size_t)ZZ * NN * 4);
    int*    idxb   = (int*)take((size_t)ZZ * CAPC * 4);
    float*  gateb  = (float*)take((size_t)ZZ * CAPC * 4);
    __bf16* wqkv_b = (__bf16*)take((size_t)DD * 3 * DD * 2);
    __bf16* wproj_b= (__bf16*)take((size_t)DD * DD * 2);
    __bf16* w1_b   = (__bf16*)take((size_t)DD * HID * 2);
    __bf16* w2_b   = (__bf16*)take((size_t)HID * DD * 2);
    __bf16* xe     = (__bf16*)take((size_t)ZZ * CAPC * DD * 2);
    __bf16* qkv    = (__bf16*)take((size_t)ZZ * CAPC * 3 * DD * 2);
    __bf16* qh     = (__bf16*)take((size_t)ZZ * CAPC * DD * 2);
    __bf16* kh     = (__bf16*)take((size_t)ZZ * CAPC * DD * 2);
    __bf16* vh     = (__bf16*)take((size_t)ZZ * CAPC * DD * 2);
    __bf16* ohb    = (__bf16*)take((size_t)ZZ * CAPC * DD * 2);
    __bf16* oflat  = (__bf16*)take((size_t)ZZ * CAPC * DD * 2);
    __bf16* ye     = (__bf16*)take((size_t)ZZ * CAPC * DD * 2);
    __bf16* hbuf   = (__bf16*)take((size_t)ZZ * CAPC * HID * 2);
    __bf16* ye2    = (__bf16*)take((size_t)ZZ * CAPC * DD * 2);

    (void)hipFuncSetAttribute((const void*)attn_fused_kernel,
                              hipFuncAttributeMaxDynamicSharedMemorySize, SMEM_ATTN);

    // weights -> bf16; residual init
    convert_bf16_kernel<<<2048, 256, 0, stream>>>(Wqkv, wqkv_b, (long)DD * 3 * DD);
    convert_bf16_kernel<<<1024, 256, 0, stream>>>(Wproj, wproj_b, (long)DD * DD);
    convert_bf16_kernel<<<2048, 256, 0, stream>>>(W1, w1_b, (long)DD * HID);
    convert_bf16_kernel<<<2048, 256, 0, stream>>>(W2, w2_b, (long)HID * DD);
    copy_f32_kernel<<<4096, 256, 0, stream>>>(x, out, (long)BB * NN * DD);

    // router + expert-choice top-CAP
    router_kernel<<<(BB * NN) / 8, 256, 0, stream>>>(x, Wr, probs);
    topk_kernel<<<ZZ, 256, 0, stream>>>(probs, idxb, gateb);

    // attention branch
    gather_ln_kernel<<<dim3(CAPC, ZZ), 256, 0, stream>>>(x, idxb, ln1g, ln1b, xe);
    gemm_bf16_kernel<0><<<dim3(18, 4, ZZ), 256, 0, stream>>>(xe, wqkv_b, nullptr, qkv,
                                                             CAPC, 3 * DD, DD, DD, DD);
    reorder_qkv_kernel<<<4096, 256, 0, stream>>>(qkv, qh, kh, vh);
    attn_fused_kernel<<<dim3(8, Z2), 256, SMEM_ATTN, stream>>>(qh, kh, vh, ohb);
    reorder_o_kernel<<<4096, 256, 0, stream>>>(ohb, oflat);
    gemm_bf16_kernel<1><<<dim3(6, 4, ZZ), 256, 0, stream>>>(oflat, wproj_b, bproj, ye,
                                                            CAPC, DD, DD, DD, DD);
    scatter_kernel<<<dim3(CAPC, ZZ), 256, 0, stream>>>(ye, idxb, gateb, out);

    // MLP branch
    gather_ln_kernel<<<dim3(CAPC, ZZ), 256, 0, stream>>>(out, idxb, ln2g, ln2b, xe);
    gemm_bf16_kernel<2><<<dim3(24, 4, ZZ), 256, 0, stream>>>(xe, w1_b, b1, hbuf,
                                                             CAPC, HID, DD, HID, HID);
    gemm_bf16_kernel<1><<<dim3(6, 4, ZZ), 256, 0, stream>>>(hbuf, w2_b, b2, ye2,
                                                            CAPC, DD, HID, DD, DD);
    scatter_kernel<<<dim3(CAPC, ZZ), 256, 0, stream>>>(ye2, idxb, gateb, out);
    (void)in_sizes; (void)n_in; (void)out_size; (void)ws_size;
}
